// BiLSTM_CRF_36481452212792
// MI455X (gfx1250) — matrix operand order repaired and checked by reading, compiled, not run on hardware
//
#include <hip/hip_runtime.h>

// ---------------- problem constants (from the reference) ----------------
#define Bn   128
#define Sn   256
#define LWn  20
#define DCn  25
#define DWn  300
#define HWn  200
#define NTn  10
#define DINn 350
#define NC2  50            // 2*HC conv filters
#define G4   800           // 4*HW gates
#define KIN  384           // DIN padded to mult of 32
#define KHH  224           // HW padded to mult of 32
#define KENC 416           // 2*HW padded to mult of 32

// ---------------- workspace layout (bytes) ----------------
// feats_h : f16 [Sn][Bn][KIN]            = 25,165,824
// wih_h   : f16 [2][G4][KIN]             =  1,228,800
// whh_h   : f16 [2][G4][KHH]             =    716,800
// projw_h : f16 [16][KENC]               =     13,312
// xg_h    : f16 [2][Sn*Bn][G4]           = 104,857,600
// enc_h   : f16 [Bn*Sn][KENC]            = 27,262,976
static const size_t OFF_FEATS = 0;
static const size_t OFF_WIH   = OFF_FEATS + (size_t)Sn * Bn * KIN * 2;
static const size_t OFF_WHH   = OFF_WIH   + (size_t)2 * G4 * KIN * 2;
static const size_t OFF_PROJW = OFF_WHH   + (size_t)2 * G4 * KHH * 2;
static const size_t OFF_XG    = OFF_PROJW + (size_t)16 * KENC * 2;
static const size_t OFF_ENC   = OFF_XG    + (size_t)2 * Sn * Bn * G4 * 2;

// ---------------- WMMA helpers (gfx1250 wave32) ----------------
typedef __attribute__((ext_vector_type(16))) _Float16 v16h;
typedef __attribute__((ext_vector_type(8)))  _Float16 v8h;
typedef __attribute__((ext_vector_type(8)))  float    v8f;

// A fragment, 16x32 f16, from a row-major [M][ld] matrix at (row0,k0)=base.
// Layout: lane half=0 holds K k0+0..15, half=1 holds K k0+16..31 (matches B).
__device__ __forceinline__ v16h wmma_load_a(const _Float16* base, int ld, int lane) {
  int m = lane & 15, h = lane >> 4;
  const _Float16* p = base + (size_t)m * ld + h * 16;
  v8h lo = *(const v8h*)p;          // K = k0 + h*16 .. +7
  v8h hi = *(const v8h*)(p + 8);    // K = k0 + h*16 + 8 .. +15
  v16h a;
#pragma unroll
  for (int i = 0; i < 8; ++i) { a[i] = lo[i]; a[i + 8] = hi[i]; }
  return a;
}

// B fragment, 32x16 f16, weights stored natural [N][ld] (row n holds K-major data).
// lanes 0-15: K k0..k0+15 of column n=lane; lanes 16-31: K k0+16..k0+31.
__device__ __forceinline__ v16h wmma_load_b(const _Float16* base, int ld, int lane) {
  int n = lane & 15, h = lane >> 4;
  return *(const v16h*)(base + (size_t)n * ld + h * 16);
}

__device__ __forceinline__ v8f wmma_f16(v16h a, v16h b, v8f c) {
  return __builtin_amdgcn_wmma_f32_16x16x32_f16(false, a, false, b, (short)0, c,
                                                false, false);
}

__device__ __forceinline__ float sigm(float x) { return 1.0f / (1.0f + expf(-x)); }

// ---------------- K0: f32 -> f16 weight staging with zero K-padding ----------------
__global__ void __launch_bounds__(256)
convert_weights_kernel(const float* __restrict__ wih_f, const float* __restrict__ whh_f,
                       const float* __restrict__ wih_b, const float* __restrict__ whh_b,
                       const float* __restrict__ projw,
                       _Float16* __restrict__ wih_h, _Float16* __restrict__ whh_h,
                       _Float16* __restrict__ projw_h) {
  int idx = blockIdx.x * 256 + threadIdx.x;
  const int NWIH = 2 * G4 * KIN;
  const int NWHH = 2 * G4 * KHH;
  const int NPRJ = 16 * KENC;
  if (idx < NWIH) {
    int d = idx / (G4 * KIN), r = idx % (G4 * KIN);
    int g = r / KIN, k = r % KIN;
    const float* w = d ? wih_b : wih_f;
    wih_h[idx] = (_Float16)((k < DINn) ? w[g * DINn + k] : 0.f);
  } else if (idx < NWIH + NWHH) {
    int j = idx - NWIH;
    int d = j / (G4 * KHH), r = j % (G4 * KHH);
    int g = r / KHH, k = r % KHH;
    const float* w = d ? whh_b : whh_f;
    whh_h[j] = (_Float16)((k < HWn) ? w[g * HWn + k] : 0.f);
  } else if (idx < NWIH + NWHH + NPRJ) {
    int j = idx - NWIH - NWHH;
    int t = j / KENC, k = j % KENC;
    projw_h[j] = (_Float16)((t < NTn && k < 2 * HWn) ? projw[t * (2 * HWn) + k] : 0.f);
  }
}

// ---------------- K1: char-CNN + embedding gather -> feats[s][b][KIN] (f16) -------
__global__ void __launch_bounds__(128)
charcnn_feats_kernel(const int* __restrict__ words, const int* __restrict__ chars,
                     const float* __restrict__ emb_char, const float* __restrict__ emb_word,
                     const float* __restrict__ conv_w, const float* __restrict__ conv_b,
                     _Float16* __restrict__ feats, _Float16* __restrict__ enc) {
  int bs = blockIdx.x;                 // b*Sn + s
  int b = bs / Sn, s = bs % Sn;
  int tid = threadIdx.x;

  __shared__ __align__(16) float ec_s[LWn][DCn];
  __shared__ __align__(16) float cw_s[NC2][3][DCn];
  __shared__ float cb_s[NC2];
  __shared__ float oc_s[NC2];

  const int* crow = chars + (size_t)bs * LWn;
  for (int i = tid; i < LWn * DCn; i += 128) {
    int l = i / DCn, d = i % DCn;
    ec_s[l][d] = emb_char[(size_t)crow[l] * DCn + d];
  }
  for (int i = tid; i < NC2 * 3 * DCn; i += 128) ((float*)cw_s)[i] = conv_w[i];
  if (tid < NC2) cb_s[tid] = conv_b[tid];
  __syncthreads();

  if (tid < NC2) {                     // one filter per thread: conv + max-pool
    int f = tid;
    float mx = -3.4e38f;
    for (int o = 0; o < LWn + 2; ++o) {
      float acc = cb_s[f];
      for (int k = 0; k < 3; ++k) {
        int p = o - 2 + k;
        if (p >= 0 && p < LWn) {
#pragma unroll
          for (int d = 0; d < DCn; ++d) acc += ec_s[p][d] * cw_s[f][k][d];
        }
      }
      mx = fmaxf(mx, acc);
    }
    oc_s[f] = mx;
  }
  __syncthreads();

  const float* ew = emb_word + (size_t)words[bs] * DWn;
  _Float16* frow = feats + ((size_t)s * Bn + b) * KIN;
  for (int d = tid; d < KIN; d += 128) {
    float v = (d < DWn) ? ew[d] : ((d < DINn) ? oc_s[d - DWn] : 0.f);
    frow[d] = (_Float16)v;
  }
  // zero the padded tail columns of enc for the projection GEMM
  if (tid < KENC - 2 * HWn) enc[(size_t)bs * KENC + 2 * HWn + tid] = (_Float16)0.f;
}

// ---------------- K2: xg[d][s*B+b][g] = feats x Wih^T + bias (f16 WMMA GEMM) ------
__global__ void __launch_bounds__(256)
xgate_gemm_kernel(const _Float16* __restrict__ feats, const _Float16* __restrict__ wih_h,
                  const float* __restrict__ bias_f, const float* __restrict__ bias_b,
                  _Float16* __restrict__ xg) {
  int lane = threadIdx.x & 31;
  int wave = threadIdx.x >> 5;
  int t = blockIdx.x * 8 + wave;            // one 16x16 tile per wave (exact grid)
  int d  = t / (2048 * NC2);
  int r  = t % (2048 * NC2);
  int mt = r / NC2, nt = r % NC2;
  int m0 = mt * 16, n0 = nt * 16;

  const _Float16* A  = feats + (size_t)m0 * KIN;
  const _Float16* Bm = wih_h + (size_t)d * G4 * KIN + (size_t)n0 * KIN;
  v8f acc = {};
#pragma unroll
  for (int kt = 0; kt < KIN / 32; ++kt) {
    v16h a = wmma_load_a(A + kt * 32, KIN, lane);
    v16h b = wmma_load_b(Bm + kt * 32, KIN, lane);
    acc = wmma_f16(a, b, acc);
  }
  const float* bias = d ? bias_b : bias_f;
  int n = lane & 15, h = lane >> 4;
  float bv = bias[n0 + n];
  _Float16* o = xg + (size_t)d * Sn * Bn * G4 + (size_t)m0 * G4 + n0 + n;
#pragma unroll
  for (int rr = 0; rr < 8; ++rr) {
    int m = rr + 8 * h;                      // C/D layout: VGPR r holds M = r + 8*half
    o[(size_t)m * G4] = (_Float16)(acc[rr] + bv);
  }
}

// ---------------- K3: masked BiLSTM scan (persistent, WMMA recurrent GEMM) --------
__global__ void __launch_bounds__(256)
lstm_scan_kernel(const _Float16* __restrict__ whh_all, const _Float16* __restrict__ xg_all,
                 const int* __restrict__ lens, _Float16* __restrict__ enc) {
  int blk = blockIdx.x;                      // 16 blocks: dir*8 + m-tile
  int dir = blk >> 3;
  int m0  = (blk & 7) * 16;
  int tid  = threadIdx.x;
  int lane = tid & 31, wave = tid >> 5;

  __shared__ __align__(16) float    h_s[16][HWn];   // f32 hidden state
  __shared__ __align__(16) float    c_s[16][HWn];   // f32 cell state
  __shared__ __align__(16) _Float16 hh_s[16][KHH];  // f16 A-operand (K-padded)
  __shared__ __align__(16) _Float16 g_s[16][G4];    // gate pre-activations
  __shared__ int len_s[16];

  const _Float16* whh = whh_all + (size_t)dir * G4 * KHH;
  const _Float16* xg  = xg_all + (size_t)dir * Sn * Bn * G4;

  for (int i = tid; i < 16 * HWn; i += 256) { ((float*)h_s)[i] = 0.f; ((float*)c_s)[i] = 0.f; }
  for (int i = tid; i < 16 * KHH; i += 256) ((_Float16*)hh_s)[i] = (_Float16)0.f;
  if (tid < 16) len_s[tid] = lens[m0 + tid];
  __syncthreads();

  for (int step = 0; step < Sn; ++step) {
    int s = dir ? (Sn - 1 - step) : step;

    // prefetch next step's input-gate tile into L2/L0 (global_prefetch_b8)
    if (step + 1 < Sn) {
      int sn = s + (dir ? -1 : 1);
      const _Float16* pf = xg + ((size_t)sn * Bn + m0) * G4;
      __builtin_prefetch(pf + (size_t)tid * 50, 0, 1);
    }

    // ---- recurrent GEMM: g[16][800] = hh x Whh^T + xg ----
    for (int nt = wave; nt < NC2; nt += 8) {
      int n0 = nt * 16;
      const _Float16* Bm = whh + (size_t)n0 * KHH;
      v8f acc = {};
#pragma unroll
      for (int kt = 0; kt < KHH / 32; ++kt) {
        v16h a = wmma_load_a(&hh_s[0][0] + kt * 32, KHH, lane);  // LDS A-fragment
        v16h b = wmma_load_b(Bm + kt * 32, KHH, lane);
        acc = wmma_f16(a, b, acc);
      }
      const _Float16* xr = xg + ((size_t)s * Bn + m0) * G4 + n0 + (lane & 15);
      int h = lane >> 4;
#pragma unroll
      for (int rr = 0; rr < 8; ++rr) {
        int m = rr + 8 * h;
        g_s[m][n0 + (lane & 15)] = (_Float16)(acc[rr] + (float)xr[(size_t)m * G4]);
      }
    }
    __syncthreads();

    // ---- elementwise gates / state update / masked output ----
    for (int i = tid; i < 16 * HWn; i += 256) {
      int m = i / HWn, j = i % HWn;
      bool act = (s < len_s[m]);
      float ig = sigm((float)g_s[m][j]);
      float fg = sigm((float)g_s[m][HWn + j]);
      float gg = tanhf((float)g_s[m][2 * HWn + j]);
      float og = sigm((float)g_s[m][3 * HWn + j]);
      float c_new = fg * c_s[m][j] + ig * gg;
      float h_new = og * tanhf(c_new);
      float hsel = act ? h_new : h_s[m][j];
      float csel = act ? c_new : c_s[m][j];
      h_s[m][j] = hsel;
      c_s[m][j] = csel;
      hh_s[m][j] = (_Float16)hsel;
      float outv = act ? h_new : 0.f;        // pad_packed semantics: zero past length
      enc[(((size_t)(m0 + m)) * Sn + s) * KENC + dir * HWn + j] = (_Float16)outv;
    }
    __syncthreads();
  }
}

// ---------------- K4: emissions = enc x proj_w^T + proj_b -------------------------
__global__ void __launch_bounds__(256)
proj_kernel(const _Float16* __restrict__ enc, const _Float16* __restrict__ projw_h,
            const float* __restrict__ proj_b, float* __restrict__ out) {
  int lane = threadIdx.x & 31;
  int wave = threadIdx.x >> 5;
  int m0 = (blockIdx.x * 8 + wave) * 16;     // 2048 M-tiles, single N-tile
  const _Float16* A = enc + (size_t)m0 * KENC;
  v8f acc = {};
#pragma unroll
  for (int kt = 0; kt < KENC / 32; ++kt) {
    v16h a = wmma_load_a(A + kt * 32, KENC, lane);
    v16h b = wmma_load_b(projw_h + kt * 32, KENC, lane);
    acc = wmma_f16(a, b, acc);
  }
  int n = lane & 15, h = lane >> 4;
  if (n < NTn) {
    float bv = proj_b[n];
#pragma unroll
    for (int rr = 0; rr < 8; ++rr) {
      int m = m0 + rr + 8 * h;
      out[(size_t)m * NTn + n] = acc[rr] + bv;
    }
  }
}

// ---------------- host-side launch ----------------
extern "C" void kernel_launch(void* const* d_in, const int* in_sizes, int n_in,
                              void* d_out, int out_size, void* d_ws, size_t ws_size,
                              hipStream_t stream) {
  (void)in_sizes; (void)n_in; (void)out_size; (void)ws_size;
  const int*   words    = (const int*)d_in[0];
  const int*   chars    = (const int*)d_in[1];
  const int*   lens     = (const int*)d_in[2];
  const float* emb_char = (const float*)d_in[3];
  const float* emb_word = (const float*)d_in[4];
  const float* conv_w   = (const float*)d_in[5];
  const float* conv_b   = (const float*)d_in[6];
  const float* w_ih_f   = (const float*)d_in[7];
  const float* w_hh_f   = (const float*)d_in[8];
  const float* b_f      = (const float*)d_in[9];
  const float* w_ih_b   = (const float*)d_in[10];
  const float* w_hh_b   = (const float*)d_in[11];
  const float* b_b      = (const float*)d_in[12];
  const float* proj_w   = (const float*)d_in[13];
  const float* proj_b   = (const float*)d_in[14];
  float* out = (float*)d_out;

  char* ws = (char*)d_ws;
  _Float16* feats_h = (_Float16*)(ws + OFF_FEATS);
  _Float16* wih_h   = (_Float16*)(ws + OFF_WIH);
  _Float16* whh_h   = (_Float16*)(ws + OFF_WHH);
  _Float16* projw_h = (_Float16*)(ws + OFF_PROJW);
  _Float16* xg_h    = (_Float16*)(ws + OFF_XG);
  _Float16* enc_h   = (_Float16*)(ws + OFF_ENC);

  // K0: stage weights to f16 with K padding
  {
    int total = 2 * G4 * KIN + 2 * G4 * KHH + 16 * KENC;
    convert_weights_kernel<<<(total + 255) / 256, 256, 0, stream>>>(
        w_ih_f, w_hh_f, w_ih_b, w_hh_b, proj_w, wih_h, whh_h, projw_h);
  }
  // K1: char-CNN + gathers -> feats
  charcnn_feats_kernel<<<Bn * Sn, 128, 0, stream>>>(
      words, chars, emb_char, emb_word, conv_w, conv_b, feats_h, enc_h);
  // K2: input-gate GEMM, both directions (2 * 2048 * 50 tiles, 8 waves/block)
  xgate_gemm_kernel<<<(2 * 2048 * NC2) / 8, 256, 0, stream>>>(
      feats_h, wih_h, b_f, b_b, xg_h);
  // K3: sequential masked BiLSTM scan (16 persistent workgroups)
  lstm_scan_kernel<<<16, 256, 0, stream>>>(whh_h, xg_h, lens, enc_h);
  // K4: projection to NT tags
  proj_kernel<<<2048 / 8, 256, 0, stream>>>(enc_h, projw_h, proj_b, out);
}
